// ZigMa_16870631538732
// MI455X (gfx1250) — compile-verified, matrix operand order checked by reading, not run on hardware
//
#include <hip/hip_runtime.h>
#include <hip/hip_bf16.h>
#include <math.h>

// ---------------------------------------------------------------------------
// ZigMa forward for MI455X (gfx1250).
// Dense projections: v_wmma_f32_16x16x32_bf16 fed by pre-converted bf16
// operands (pure global_load_b128 + wmma inner loop, zero in-loop converts).
// Scan / LN / conv / gates: fp32 VALU (reference numerics).
// ---------------------------------------------------------------------------

#define BB    4
#define LL    1024
#define DM    768
#define DI    1536
#define DS    16
#define DTR   48
#define NPROJ 80          // DT_RANK + 2*D_STATE
#define NDEPTH 4

typedef __attribute__((ext_vector_type(16))) __bf16 v16bf;
typedef __attribute__((ext_vector_type(8)))  __bf16 v8bf;
typedef __attribute__((ext_vector_type(8)))  float  v8f;

__device__ __forceinline__ float silu_f(float x) {
    return x / (1.f + __expf(-x));
}
__device__ __forceinline__ __bf16 to_bf16(float f) {
    return (__bf16)f;
}

// 16-byte packed bf16 fragment loads (ISA 16-bit A/B striping: per lane,
// 8 contiguous K at kb and 8 contiguous K at kb+16).
__device__ __forceinline__ v16bf load_frag(const __bf16* __restrict__ row, int kb) {
    v8bf lo = *(const v8bf*)(row + kb);
    v8bf hi = *(const v8bf*)(row + kb + 16);
    return __builtin_shufflevector(lo, hi, 0,1,2,3,4,5,6,7,8,9,10,11,12,13,14,15);
}
__device__ __forceinline__ v16bf load_frag_guard(const __bf16* __restrict__ row,
                                                 int kb, int K) {
    v16bf f;
    #pragma unroll
    for (int j = 0; j < 8; ++j) {
        int k1 = kb + j, k2 = kb + 16 + j;
        f[j]     = (k1 < K) ? row[k1] : (__bf16)0.f;
        f[8 + j] = (k2 < K) ? row[k2] : (__bf16)0.f;
    }
    return f;
}

// ---------------------------------------------------------------------------
// WMMA GEMM:  C[m][n] = epi( sum_k A[m][k] * W[n][k] ),  A/W bf16, acc fp32.
// One wave computes a 16 x (16*NT) tile (A fragment reused NT times).
// ---------------------------------------------------------------------------
#define EPI_NONE     0
#define EPI_DUAL     1   // fp32 store + bf16 side copy
#define EPI_SOFTPLUS 2   // + bias, softplus
#define EPI_SCATTER  3   // + residual, inverse-permutation scatter

template<int EPI, int NT, bool KGUARD>
__global__ __launch_bounds__(256)
void wmma_gemm_kernel(const __bf16* __restrict__ A, int lda,
                      const __bf16* __restrict__ W, int ldb,
                      float* __restrict__ C, int ldc,
                      __bf16* __restrict__ Cbf,
                      int M, int N, int K,
                      const float* __restrict__ bias,
                      const int*   __restrict__ order,
                      float*       __restrict__ scat)
{
    const int lane  = threadIdx.x & 31;
    const int wave  = threadIdx.x >> 5;
    const int wpb   = blockDim.x >> 5;
    const int tileM = blockIdx.x * 16;
    const int tileN = (blockIdx.y * wpb + wave) * (16 * NT);
    const int half  = lane >> 4;       // K-half select (ISA 16-bit A/B layout)
    const int mr    = lane & 15;       // A row within tile
    const int nr    = lane & 15;       // B col within tile

    v8f acc[NT];
    const v8f vzero = {0.f,0.f,0.f,0.f,0.f,0.f,0.f,0.f};
    #pragma unroll
    for (int t = 0; t < NT; ++t) acc[t] = vzero;

    const __bf16* arow = A + (size_t)(tileM + mr) * lda;

    for (int k0 = 0; k0 < K; k0 += 32) {
        const int kb = k0 + half * 8;
        if (k0 + 32 < K) __builtin_prefetch(arow + k0 + 32, 0, 1);  // global_prefetch_b8

        v16bf afrag = KGUARD ? load_frag_guard(arow, kb, K)
                             : load_frag(arow, kb);
        #pragma unroll
        for (int t = 0; t < NT; ++t) {
            const __bf16* wrow = W + (size_t)(tileN + t * 16 + nr) * ldb;
            v16bf bfrag = KGUARD ? load_frag_guard(wrow, kb, K)
                                 : load_frag(wrow, kb);
            acc[t] = __builtin_amdgcn_wmma_f32_16x16x32_bf16(
                false, afrag, false, bfrag, (short)0, acc[t], false, false);
        }
    }

    // D layout: lane holds col nr, rows tileM + half*8 + j (j = VGPR index)
    #pragma unroll
    for (int t = 0; t < NT; ++t) {
        const int n = tileN + t * 16 + nr;
        const int mbase = tileM + half * 8;
        #pragma unroll
        for (int j = 0; j < 8; ++j) {
            const int m = mbase + j;
            float v = acc[t][j];
            if constexpr (EPI == EPI_SOFTPLUS) {
                v += bias[n];
                v = (v > 20.f) ? v : log1pf(__expf(v));   // softplus
                C[(size_t)m * ldc + n] = v;
            } else if constexpr (EPI == EPI_SCATTER) {
                const int b = m >> 10, l = m & 1023;
                const size_t row = (size_t)(b * LL + order[l]) * ldc;
                C[row + n] = v + scat[row + n];           // residual + inv-perm
            } else if constexpr (EPI == EPI_DUAL) {
                C[(size_t)m * ldc + n]   = v;
                Cbf[(size_t)m * ldc + n] = to_bf16(v);
            } else {
                C[(size_t)m * ldc + n] = v;
            }
        }
    }
}

// ---------------------------------------------------------------------------
// fp32 -> bf16 weight conversion (once per launch; weights then stay in L2)
// ---------------------------------------------------------------------------
__global__ void cvt_bf16_kernel(const float* __restrict__ src,
                                __bf16* __restrict__ dst, int n)
{
    int i = blockIdx.x * blockDim.x + threadIdx.x;
    if (i < n) dst[i] = to_bf16(src[i]);
}

// ---------------------------------------------------------------------------
// Elementwise / small kernels (fp32)
// ---------------------------------------------------------------------------
__global__ void patch_embed_kernel(const float* __restrict__ x,
                                   const float* __restrict__ pw,
                                   const float* __restrict__ pb,
                                   const float* __restrict__ pos,
                                   float* __restrict__ h)
{
    int idx = blockIdx.x * blockDim.x + threadIdx.x;     // B*L*DM
    int d = idx % DM, m = idx / DM;
    int b = m >> 10, l = m & 1023;
    float acc = pb[d] + pos[l * DM + d];
    #pragma unroll
    for (int c = 0; c < 4; ++c)
        acc += x[b * 4096 + c * LL + l] * pw[d * 4 + c];
    h[idx] = acc;
}

__global__ void time_mlp1_kernel(const float* __restrict__ t,
                                 const float* __restrict__ w1,
                                 const float* __restrict__ b1,
                                 float* __restrict__ hmid)
{
    int idx = blockIdx.x * blockDim.x + threadIdx.x;     // B*DM
    int d = idx % DM, b = idx / DM;
    float tv = t[b];
    float acc = b1[d];
    const float c0 = -logf(10000.f) / 128.f;
    for (int j = 0; j < 128; ++j) {
        float ang = tv * __expf(c0 * (float)j);
        acc += __cosf(ang) * w1[d * 256 + j] + __sinf(ang) * w1[d * 256 + 128 + j];
    }
    hmid[idx] = silu_f(acc);
}

__global__ void time_mlp2_kernel(const float* __restrict__ hmid,
                                 const float* __restrict__ w2,
                                 const float* __restrict__ b2,
                                 float* __restrict__ cbuf)
{
    int idx = blockIdx.x * blockDim.x + threadIdx.x;     // B*DM
    int d = idx % DM, b = idx / DM;
    float acc = b2[d];
    for (int j = 0; j < DM; ++j)
        acc += hmid[b * DM + j] * w2[d * DM + j];
    cbuf[idx] = acc;
}

__global__ void ada_kernel(const float* __restrict__ cbuf,
                           const float* __restrict__ aw,
                           const float* __restrict__ ab,
                           float* __restrict__ adasc)
{
    int idx = blockIdx.x * blockDim.x + threadIdx.x;     // B*2*DM
    int n = idx % (2 * DM), b = idx / (2 * DM);
    float acc = ab[n];
    for (int j = 0; j < DM; ++j)
        acc += silu_f(cbuf[b * DM + j]) * aw[n * DM + j];
    adasc[idx] = acc;
}

// gather by zigzag order + LayerNorm -> bf16 GEMM operand ; one block per (b,l)
__global__ __launch_bounds__(256)
void gather_ln_kernel(const float* __restrict__ h,
                      const int*   __restrict__ order,
                      const float* __restrict__ nw,
                      const float* __restrict__ nb,
                      __bf16* __restrict__ ubf)
{
    const int m = blockIdx.x;                 // b*L + l
    const int b = m >> 10, l = m & 1023;
    const float* row = h + (size_t)(b * LL + order[l]) * DM;
    __shared__ float r1[256], r2[256];
    float s = 0.f, s2 = 0.f;
    for (int d = threadIdx.x; d < DM; d += 256) {
        float v = row[d]; s += v; s2 += v * v;
    }
    r1[threadIdx.x] = s; r2[threadIdx.x] = s2;
    __syncthreads();
    for (int st = 128; st > 0; st >>= 1) {
        if (threadIdx.x < st) { r1[threadIdx.x] += r1[threadIdx.x + st];
                                r2[threadIdx.x] += r2[threadIdx.x + st]; }
        __syncthreads();
    }
    const float mean = r1[0] / (float)DM;
    const float var  = r2[0] / (float)DM - mean * mean;
    const float rs   = rsqrtf(var + 1e-5f);
    __bf16* urow = ubf + (size_t)m * DM;
    for (int d = threadIdx.x; d < DM; d += 256)
        urow[d] = to_bf16((row[d] - mean) * rs * nw[d] + nb[d]);
}

// causal depthwise conv (D_CONV=4) + bias + SiLU; writes fp32 (scan) + bf16 (GEMM)
__global__ void conv_silu_kernel(const float* __restrict__ xz,
                                 const float* __restrict__ cw,
                                 const float* __restrict__ cb,
                                 float* __restrict__ xconv,
                                 __bf16* __restrict__ xconv_bf)
{
    int idx = blockIdx.x * blockDim.x + threadIdx.x;     // B*L*DI
    int d = idx % DI, m = idx / DI;
    int b = m >> 10, l = m & 1023;
    float acc = cb[d];
    #pragma unroll
    for (int j = 0; j < 4; ++j) {
        int ls = l - 3 + j;
        if (ls >= 0)
            acc += xz[(size_t)(b * LL + ls) * (2 * DI) + d] * cw[d * 4 + j];
    }
    float v = silu_f(acc);
    xconv[(size_t)m * DI + d]    = v;
    xconv_bf[(size_t)m * DI + d] = to_bf16(v);
}

// selective scan: thread = (b, channel d); 1024 sequential steps; LDS bcast B/C
__global__ __launch_bounds__(256)
void scan_kernel(const float* __restrict__ dtv,     // softplus(dt)
                 const float* __restrict__ xconv,
                 const float* __restrict__ dbl,     // [.,80]: dt|B|C (fp32)
                 const float* __restrict__ xz,      // z at col DI+d
                 const float* __restrict__ A_log,
                 const float* __restrict__ Dp,
                 __bf16* __restrict__ ybf)          // gated output -> GEMM A
{
    const int b = blockIdx.x / (DI / 256);
    const int d = (blockIdx.x % (DI / 256)) * 256 + threadIdx.x;
    __shared__ float bc[32];
    float a_s[DS];
    #pragma unroll
    for (int s = 0; s < DS; ++s) a_s[s] = -__expf(A_log[d * DS + s]);
    const float dcoef = Dp[d];
    float st[DS];
    #pragma unroll
    for (int s = 0; s < DS; ++s) st[s] = 0.f;

    for (int l = 0; l < LL; ++l) {
        const int m = b * LL + l;
        __syncthreads();
        if (threadIdx.x < 32)
            bc[threadIdx.x] = dbl[(size_t)m * NPROJ + DTR + threadIdx.x];
        __syncthreads();
        const float dt  = dtv[(size_t)m * DI + d];
        const float x   = xconv[(size_t)m * DI + d];
        const float dtx = dt * x;
        float y = 0.f;
        #pragma unroll
        for (int s = 0; s < DS; ++s) {
            st[s] = __expf(dt * a_s[s]) * st[s] + dtx * bc[s];
            y += st[s] * bc[DS + s];
        }
        const float zv = xz[(size_t)m * (2 * DI) + DI + d];
        ybf[(size_t)m * DI + d] = to_bf16((y + dcoef * x) * silu_f(zv));
    }
}

// final AdaLN modulate + linear to C_IN ; one block per (b,l)
__global__ __launch_bounds__(256)
void final_kernel(const float* __restrict__ h,
                  const float* __restrict__ adasc,
                  const float* __restrict__ flw,
                  const float* __restrict__ flb,
                  float* __restrict__ out)
{
    const int m = blockIdx.x;
    const int b = m >> 10, l = m & 1023;
    const float* row = h + (size_t)m * DM;
    const float* shift = adasc + b * 2 * DM;
    const float* scale = shift + DM;
    __shared__ float r1[256], r2[256];
    float s = 0.f, s2 = 0.f;
    for (int d = threadIdx.x; d < DM; d += 256) {
        float v = row[d]; s += v; s2 += v * v;
    }
    r1[threadIdx.x] = s; r2[threadIdx.x] = s2;
    __syncthreads();
    for (int st = 128; st > 0; st >>= 1) {
        if (threadIdx.x < st) { r1[threadIdx.x] += r1[threadIdx.x + st];
                                r2[threadIdx.x] += r2[threadIdx.x + st]; }
        __syncthreads();
    }
    const float mean = r1[0] / (float)DM;
    const float var  = r2[0] / (float)DM - mean * mean;
    const float rs   = rsqrtf(var + 1e-6f);
    __syncthreads();

    float acc[4] = {0.f, 0.f, 0.f, 0.f};
    for (int d = threadIdx.x; d < DM; d += 256) {
        float hn = (row[d] - mean) * rs;
        hn = hn * (1.f + scale[d]) + shift[d];
        #pragma unroll
        for (int c = 0; c < 4; ++c) acc[c] += hn * flw[c * DM + d];
    }
    for (int c = 0; c < 4; ++c) {
        r1[threadIdx.x] = acc[c];
        __syncthreads();
        for (int st = 128; st > 0; st >>= 1) {
            if (threadIdx.x < st) r1[threadIdx.x] += r1[threadIdx.x + st];
            __syncthreads();
        }
        if (threadIdx.x == 0)
            out[(size_t)b * 4096 + c * LL + l] = r1[0] + flb[c];
        __syncthreads();
    }
}

// ---------------------------------------------------------------------------
extern "C" void kernel_launch(void* const* d_in, const int* in_sizes, int n_in,
                              void* d_out, int out_size, void* d_ws, size_t ws_size,
                              hipStream_t stream)
{
    (void)in_sizes; (void)n_in; (void)out_size; (void)ws_size;

    const float* x        = (const float*)d_in[0];
    const float* t        = (const float*)d_in[1];
    const float* patch_w  = (const float*)d_in[2];
    const float* patch_b  = (const float*)d_in[3];
    const float* pos      = (const float*)d_in[4];
    const float* t_w1     = (const float*)d_in[5];
    const float* t_b1     = (const float*)d_in[6];
    const float* t_w2     = (const float*)d_in[7];
    const float* t_b2     = (const float*)d_in[8];
    const float* norm_w   = (const float*)d_in[9];
    const float* norm_b   = (const float*)d_in[10];
    const float* in_pw    = (const float*)d_in[11];
    const float* conv_w   = (const float*)d_in[12];
    const float* conv_b   = (const float*)d_in[13];
    const float* x_pw     = (const float*)d_in[14];
    const float* dt_w     = (const float*)d_in[15];
    const float* dt_b     = (const float*)d_in[16];
    const float* A_log    = (const float*)d_in[17];
    const float* Dp       = (const float*)d_in[18];
    const float* out_pw   = (const float*)d_in[19];
    const float* ada_w    = (const float*)d_in[20];
    const float* ada_b    = (const float*)d_in[21];
    const float* flin_w   = (const float*)d_in[22];
    const float* flin_b   = (const float*)d_in[23];
    const int*   orders   = (const int*)d_in[24];
    float* out = (float*)d_out;

    const int M = BB * LL;                 // 4096 token rows

    // ---- workspace carve-up: fp32 region, then 16B-aligned bf16 region ----
    float* wsf = (float*)d_ws;
    float* h      = wsf;  wsf += (size_t)M * DM;        // 3.1M
    float* xz     = wsf;  wsf += (size_t)M * 2 * DI;    // 12.6M
    float* xconv  = wsf;  wsf += (size_t)M * DI;        // 6.3M
    float* dbl    = wsf;  wsf += (size_t)M * NPROJ;     // 0.33M
    float* dtv    = wsf;  wsf += (size_t)M * DI;        // 6.3M
    float* hmid   = wsf;  wsf += BB * DM;
    float* cbuf   = wsf;  wsf += BB * DM;
    float* adasc  = wsf;  wsf += BB * 2 * DM;

    __bf16* wsb = (__bf16*)wsf;
    __bf16* u_bf     = wsb;  wsb += (size_t)M * DM;
    __bf16* xconv_bf = wsb;  wsb += (size_t)M * DI;
    __bf16* dbl_bf   = wsb;  wsb += (size_t)M * NPROJ;
    __bf16* y_bf     = wsb;  wsb += (size_t)M * DI;
    __bf16* w_in_bf  = wsb;  wsb += (size_t)NDEPTH * 2 * DI * DM;
    __bf16* w_x_bf   = wsb;  wsb += (size_t)NDEPTH * NPROJ * DI;
    __bf16* w_dt_bf  = wsb;  wsb += (size_t)NDEPTH * DI * DTR;
    __bf16* w_out_bf = wsb;  wsb += (size_t)NDEPTH * DM * DI;

    // ---- one-time (per launch) weight conversion to bf16 -----------------
    {
        const int n_in_w  = NDEPTH * 2 * DI * DM;   // 9,437,184
        const int n_x_w   = NDEPTH * NPROJ * DI;    //   491,520
        const int n_dt_w  = NDEPTH * DI * DTR;      //   294,912
        const int n_out_w = NDEPTH * DM * DI;       // 4,718,592
        cvt_bf16_kernel<<<n_in_w  / 256, 256, 0, stream>>>(in_pw,  w_in_bf,  n_in_w);
        cvt_bf16_kernel<<<n_x_w   / 256, 256, 0, stream>>>(x_pw,   w_x_bf,   n_x_w);
        cvt_bf16_kernel<<<n_dt_w  / 256, 256, 0, stream>>>(dt_w,   w_dt_bf,  n_dt_w);
        cvt_bf16_kernel<<<n_out_w / 256, 256, 0, stream>>>(out_pw, w_out_bf, n_out_w);
    }

    // ---- stem + conditioning ---------------------------------------------
    patch_embed_kernel<<<(M * DM) / 256, 256, 0, stream>>>(x, patch_w, patch_b, pos, h);
    time_mlp1_kernel<<<(BB * DM) / 256, 256, 0, stream>>>(t, t_w1, t_b1, hmid);
    time_mlp2_kernel<<<(BB * DM) / 256, 256, 0, stream>>>(hmid, t_w2, t_b2, cbuf);
    ada_kernel<<<(BB * 2 * DM) / 256, 256, 0, stream>>>(cbuf, ada_w, ada_b, adasc);

    // ---- 4 ZigMa layers ---------------------------------------------------
    for (int i = 0; i < NDEPTH; ++i) {
        const int*    ord = orders   + i * LL;
        const __bf16* wi  = w_in_bf  + (size_t)i * 2 * DI * DM;
        const __bf16* wx  = w_x_bf   + (size_t)i * NPROJ * DI;
        const __bf16* wdt = w_dt_bf  + (size_t)i * DI * DTR;
        const __bf16* wo  = w_out_bf + (size_t)i * DM * DI;

        gather_ln_kernel<<<M, 256, 0, stream>>>(
            h, ord, norm_w + i * DM, norm_b + i * DM, u_bf);

        // in_proj: 4096 x 768 -> 3072   (8 waves * NT=2 -> 256 cols / block)
        wmma_gemm_kernel<EPI_NONE, 2, false><<<dim3(M / 16, 12), 256, 0, stream>>>(
            u_bf, DM, wi, DM, xz, 2 * DI, nullptr, M, 2 * DI, DM,
            nullptr, nullptr, nullptr);

        conv_silu_kernel<<<(M * DI) / 256, 256, 0, stream>>>(
            xz, conv_w + (size_t)i * DI * 4, conv_b + i * DI, xconv, xconv_bf);

        // x_proj: 4096 x 1536 -> 80    (5 waves, NT=1 -> 80 cols / block)
        //   dual store: fp32 (scan B/C) + bf16 (dt GEMM A operand)
        wmma_gemm_kernel<EPI_DUAL, 1, false><<<dim3(M / 16, 1), 160, 0, stream>>>(
            xconv_bf, DI, wx, DI, dbl, NPROJ, dbl_bf, M, NPROJ, DI,
            nullptr, nullptr, nullptr);

        // dt_proj (+bias, softplus): 4096 x 48 -> 1536 ; K tail-guarded
        wmma_gemm_kernel<EPI_SOFTPLUS, 2, true><<<dim3(M / 16, 6), 256, 0, stream>>>(
            dbl_bf, NPROJ, wdt, DTR, dtv, DI, nullptr, M, DI, DTR,
            dt_b + i * DI, nullptr, nullptr);

        // serial selective scan + D skip + SiLU(z) gate -> bf16 y
        scan_kernel<<<BB * (DI / 256), 256, 0, stream>>>(
            dtv, xconv, dbl, xz, A_log + (size_t)i * DI * DS, Dp + i * DI, y_bf);

        // out_proj + residual + inverse-permutation scatter into h (in place)
        wmma_gemm_kernel<EPI_SCATTER, 2, false><<<dim3(M / 16, 3), 256, 0, stream>>>(
            y_bf, DI, wo, DI, h, DM, nullptr, M, DM, DI,
            nullptr, ord, h);
    }

    // ---- final modulated LN + linear -------------------------------------
    final_kernel<<<M, 256, 0, stream>>>(h, adasc, flin_w, flin_b, out);
}
